// GCNNOperator_74526272520300
// MI455X (gfx1250) — matrix operand order repaired and checked by reading, not compiled, not run on hardware
//
#include <hip/hip_runtime.h>

// MI455X / gfx1250: wave32, WMMA 16x16x32 bf16 with fp32 accumulate.
//
// out[b] = A(16384x16384) @ h[b](16384x16) + bias,  h[b] = per-node x@kernel.
// A (1 GiB fp32) dominates bytes -> stream it non-temporally ONCE, convert to
// bf16 in registers, and amortize each A tile over both batches (2 WMMAs per
// A load). h (1 MB bf16) is precomputed transposed in d_ws and served from L2.

#define NN     16384
#define FF     16
#define GG     16
#define BB     2
#define KSTEP  32
#define WAVES  8          // waves per block in stage 2
#define KSLICE (NN / WAVES)

typedef __bf16 bf16x16 __attribute__((ext_vector_type(16)));
typedef float  f32x8   __attribute__((ext_vector_type(8)));
typedef float  f32x4   __attribute__((ext_vector_type(4)));

// ---------------------------------------------------------------------------
// Stage 1: h[b,n,g] = sum_f x[b,n,f] * kernel[n,f,g], stored transposed as
// bf16: hT[b][g][m]  (so stage-2 B-operand loads are 32B contiguous per lane)
// ---------------------------------------------------------------------------
__global__ __launch_bounds__(256) void gcn_stage1(
    const float* __restrict__ x,      // [B, N, F]
    const float* __restrict__ W,      // [N, F, G]
    __bf16* __restrict__ hT)          // [B, G, N] bf16
{
    const int gid = blockIdx.x * 256 + threadIdx.x;   // 0 .. B*N-1
    const int b = gid >> 14;
    const int n = gid & (NN - 1);

    const float* xp = x + (((size_t)b << 14) + n) * FF;
    float xs[FF];
#pragma unroll
    for (int f = 0; f < FF; ++f) xs[f] = xp[f];

    float acc[GG];
#pragma unroll
    for (int g = 0; g < GG; ++g) acc[g] = 0.0f;

    const float* wp = W + (size_t)n * (FF * GG);      // kernel[n] is contiguous 1KB
#pragma unroll
    for (int f = 0; f < FF; ++f) {
#pragma unroll
        for (int g = 0; g < GG; ++g)
            acc[g] = fmaf(xs[f], wp[f * GG + g], acc[g]);
    }

#pragma unroll
    for (int g = 0; g < GG; ++g)
        hT[(((size_t)b * GG + g) << 14) + n] = (__bf16)acc[g];
}

// ---------------------------------------------------------------------------
// Stage 2: one block per 16-row band of A. 8 waves split K=16384; each wave
// accumulates BOTH batch tiles from one A-register load (v_wmma_f32_16x16x32_bf16),
// then LDS cross-wave reduction + bias + store.
// ---------------------------------------------------------------------------
__global__ __launch_bounds__(256) void gcn_stage2(
    const float*  __restrict__ A,     // [N, N]
    const __bf16* __restrict__ hT,    // [B, G, N] bf16
    const float*  __restrict__ bias,  // [N, G]
    float*        __restrict__ out)   // [B, N, G]
{
    const int lane  = threadIdx.x & 31;
    const int w     = threadIdx.x >> 5;       // wave id 0..7
    const int r     = lane & 15;              // A row within band / B column g
    const int khalf = lane >> 4;              // 0: low-K half, 1: high-K half

    // A-operand (16x32 bf16): lane(0-15)=row M, K {0..15};
    // lanes 16-31: M=lane&15, K {16..31}.
    const float* arow = A + (size_t)(blockIdx.x * 16 + r) * NN + khalf * 16;

    // B-operand (32x16 bf16): lane column g = lane&15; 16 contiguous K values.
    const __bf16* hb0 = hT + ((size_t)r << 14) + khalf * 16;            // b = 0
    const __bf16* hb1 = hb0 + ((size_t)GG << 14);                       // b = 1

    f32x8 c0 = {};
    f32x8 c1 = {};

    const int kbeg = w * KSLICE;
#pragma unroll 4
    for (int k = kbeg; k < kbeg + KSLICE; k += KSTEP) {
        const float* ap = arow + k;
        // A is streamed exactly once -> non-temporal, keep L2 for hT.
        f32x4 a0 = __builtin_nontemporal_load((const f32x4*)(ap));
        f32x4 a1 = __builtin_nontemporal_load((const f32x4*)(ap + 4));
        f32x4 a2 = __builtin_nontemporal_load((const f32x4*)(ap + 8));
        f32x4 a3 = __builtin_nontemporal_load((const f32x4*)(ap + 12));
        __builtin_prefetch(ap + 1024, 0, 0);   // global_prefetch_b8, 4KB ahead

        bf16x16 av;
#pragma unroll
        for (int i = 0; i < 4; ++i) {
            av[i]      = (__bf16)a0[i];
            av[4 + i]  = (__bf16)a1[i];
            av[8 + i]  = (__bf16)a2[i];
            av[12 + i] = (__bf16)a3[i];
        }

        bf16x16 b0 = *(const bf16x16*)(hb0 + k);   // 32B, L2-resident
        bf16x16 b1 = *(const bf16x16*)(hb1 + k);

        c0 = __builtin_amdgcn_wmma_f32_16x16x32_bf16(
                 false, av, false, b0, (short)0, c0, false, false);
        c1 = __builtin_amdgcn_wmma_f32_16x16x32_bf16(
                 false, av, false, b1, (short)0, c1, false, false);
    }

    // Cross-wave reduction of the two 16x16 fp32 tiles.
    __shared__ float red[WAVES][BB][32][8];
#pragma unroll
    for (int j = 0; j < 8; ++j) {
        red[w][0][lane][j] = c0[j];
        red[w][1][lane][j] = c1[j];
    }
    __syncthreads();

#pragma unroll
    for (int s = threadIdx.x; s < BB * 32 * 8; s += 256) {
        const int tile  = s >> 8;          // batch
        const int lane2 = (s >> 3) & 31;
        const int j     = s & 7;
        float acc = 0.0f;
#pragma unroll
        for (int ww = 0; ww < WAVES; ++ww) acc += red[ww][tile][lane2][j];

        // C/D layout: M = j + 8*(lane>=16), g = lane & 15
        const int M    = j + ((lane2 >> 4) << 3);
        const int g    = lane2 & 15;
        const int nrow = blockIdx.x * 16 + M;
        acc += bias[nrow * GG + g];
        out[((size_t)tile << 18) + (size_t)nrow * GG + g] = acc;
    }
}

// ---------------------------------------------------------------------------
extern "C" void kernel_launch(void* const* d_in, const int* in_sizes, int n_in,
                              void* d_out, int out_size, void* d_ws, size_t ws_size,
                              hipStream_t stream) {
    const float* x    = (const float*)d_in[0];   // [2, 16384, 16]
    const float* A    = (const float*)d_in[1];   // [16384, 16384]
    const float* W    = (const float*)d_in[2];   // [16384, 16, 16]
    const float* bias = (const float*)d_in[3];   // [16384, 16]
    float*       out  = (float*)d_out;           // [2, 16384, 16]
    __bf16*      hT   = (__bf16*)d_ws;           // [2, 16, 16384] bf16 = 1 MB

    gcn_stage1<<<(BB * NN) / 256, 256, 0, stream>>>(x, W, hT);
    gcn_stage2<<<NN / 16, 256, 0, stream>>>(A, hT, bias, out);
}
